// SwinTransformerBlock_90340342104423
// MI455X (gfx1250) — matrix operand order repaired and checked by reading, not compiled, not run on hardware
//
#include <hip/hip_runtime.h>
#include <hip/hip_bf16.h>

#define DIM    192
#define HEADS  6
#define HD     32
#define WS_    7
#define SS_    3
#define NTOK   49
#define NPAD   64
#define IMG    56
#define BATCH  32
#define NWINDOWS 2048            // 32 * 64 windows
#define MLPH   768
#define TOKENS (BATCH*IMG*IMG)   // 100352

typedef _Float16 h16;
typedef __attribute__((ext_vector_type(16))) _Float16 v16h;
typedef __attribute__((ext_vector_type(8)))  float    v8f;
typedef __attribute__((ext_vector_type(4)))  unsigned int u32x4;
typedef __attribute__((ext_vector_type(4)))  int i32x4;
typedef __attribute__((address_space(1))) i32x4 gi32x4;   // global v4i32
typedef __attribute__((address_space(3))) i32x4 li32x4;   // LDS v4i32

union V16HQ { v16h v; u32x4 q[2]; };

#if defined(__has_builtin)
#if __has_builtin(__builtin_amdgcn_global_load_async_to_lds_b128)
#define ASYNC_LDS 1
#endif
#endif

// direct global->LDS async copy of 16 bytes per lane (ASYNCcnt path);
// falls back to VGPR round-trip if the builtin is not declared.
__device__ __forceinline__ void async_copy16(const h16* g, h16* l) {
#ifdef ASYNC_LDS
  __builtin_amdgcn_global_load_async_to_lds_b128((gi32x4*)g, (li32x4*)l, 0, 0);
#else
  *(u32x4*)l = *(const u32x4*)g;
#endif
}
__device__ __forceinline__ void async_wait() {
#ifdef ASYNC_LDS
#if __has_builtin(__builtin_amdgcn_s_wait_asynccnt)
  __builtin_amdgcn_s_wait_asynccnt(0);
#else
  asm volatile("s_wait_asynccnt 0x0" ::: "memory");
#endif
#endif
}

// ---- WMMA operand loaders -------------------------------------------------
// A (row-major [M][K] in LDS): lane m holds K 0-15, lane m+16 holds 16-31
__device__ __forceinline__ v16h lds_load_a(const h16* s, int stride, int mb, int kb, int lane) {
  int m  = mb + (lane & 15);
  int hi = (lane >> 4) & 1;
  const h16* p = s + m * stride + kb + hi * 16;
  V16HQ r;
  r.q[0] = *(const u32x4*)(p);
  r.q[1] = *(const u32x4*)(p + 8);
  return r.v;
}
// B given as B^T row-major [N][K] in LDS: lane n holds K 0-15, lane n+16 holds 16-31
__device__ __forceinline__ v16h lds_load_bt(const h16* s, int stride, int nb, int kb, int lane) {
  int n  = nb + (lane & 15);
  int hi = (lane >> 4) & 1;
  const h16* p = s + n * stride + kb + hi * 16;
  V16HQ r;
  r.q[0] = *(const u32x4*)(p);
  r.q[1] = *(const u32x4*)(p + 8);
  return r.v;
}
// Pre-swizzled weight in global: per (otile,kchunk) 32 lanes x 16 contiguous halves
__device__ __forceinline__ v16h gbl_load_w(const h16* w, int nKC, int ot, int kc, int lane) {
  const h16* p = w + (((ot * nKC) + kc) * 32 + lane) * 16;
  V16HQ r;
  r.q[0] = *(const u32x4*)(p);
  r.q[1] = *(const u32x4*)(p + 8);
  return r.v;
}
__device__ __forceinline__ v8f wmma16(v16h a, v16h b, v8f c) {
  return __builtin_amdgcn_wmma_f32_16x16x32_f16(false, a, false, b, (short)0, c, false, false);
}

// ---- weight convert + swizzle to WMMA B layout ---------------------------
__global__ void k_wconv(const float* __restrict__ src, h16* __restrict__ dst,
                        int nOT, int nKC, int C) {
  int f = blockIdx.x * 256 + threadIdx.x;
  int total = nOT * nKC * 512;
  if (f >= total) return;
  int e  = f & 15;
  int l  = (f >> 4) & 31;
  int kc = (f >> 9) % nKC;
  int ot = f / (512 * nKC);
  int o  = ot * 16 + (l & 15);
  int k  = kc * 32 + ((l >> 4) & 1) * 16 + e;
  dst[f] = (h16)src[o * C + k];
}

// ---- CPB-MLP relative position bias: rpb[h][i][j] = 14*sigmoid(mlp(table)[rpi]) ----
__global__ void k_bias(const float* __restrict__ w1, const float* __restrict__ b1,
                       const float* __restrict__ w2, float* __restrict__ rpb) {
  __shared__ float hbt[169 * HEADS];
  int tid = threadIdx.x;
  const float il7 = 1.0f / log2f(7.0f);
  for (int p = tid; p < 169 * HEADS; p += 256) {
    int e = p / HEADS, h = p % HEADS;
    int i = e / 13, j = e % 13;
    float r0 = (float)(i - 6) * (7.0f / 6.0f);
    float r1 = (float)(j - 6) * (7.0f / 6.0f);
    float t0 = (r0 >= 0.f ? 1.f : -1.f) * log2f(fabsf(r0) + 1.f) * il7;
    float t1 = (r1 >= 0.f ? 1.f : -1.f) * log2f(fabsf(r1) + 1.f) * il7;
    float acc = 0.f;
    for (int q = 0; q < 512; ++q) {
      float hv = w1[q * 2] * t0 + w1[q * 2 + 1] * t1 + b1[q];
      hv = hv > 0.f ? hv : 0.01f * hv;         // leaky relu
      acc += hv * w2[h * 512 + q];
    }
    hbt[p] = acc;
  }
  __syncthreads();
  for (int p = tid; p < HEADS * NTOK * NTOK; p += 256) {
    int h = p / (NTOK * NTOK);
    int r = (p / NTOK) % NTOK;
    int c = p % NTOK;
    int d0 = (r / 7) - (c / 7) + 6;
    int d1 = (r % 7) - (c % 7) + 6;
    float v = hbt[(d0 * 13 + d1) * HEADS + h];
    rpb[p] = 14.0f / (1.0f + expf(-v));
  }
}

// ---- shift + window partition -> f16 [2048][64][192], rows 49..63 zero ----
__global__ void k_pack(const float* __restrict__ x, h16* __restrict__ xwh) {
  int win = blockIdx.x;
  int b = win >> 6, widx = win & 63;
  int wh = widx >> 3, ww = widx & 7;
  for (int i = threadIdx.x; i < NPAD * DIM; i += 256) {
    int n = i / DIM, c = i % DIM;
    float v = 0.f;
    if (n < NTOK) {
      int gh = (wh * WS_ + n / WS_ + SS_) % IMG;
      int gw = (ww * WS_ + n % WS_ + SS_) % IMG;
      v = x[((b * IMG + gh) * IMG + gw) * DIM + c];
    }
    xwh[win * (NPAD * DIM) + i] = (h16)v;
  }
}

// ---- fused QKV + cosine attention + softmax + probs@V (per window, head) ----
__global__ __launch_bounds__(256) void k_attn(
    const h16* __restrict__ xwh, const h16* __restrict__ wq,
    const float* __restrict__ qkv_b, const float* __restrict__ logit_scale,
    const float* __restrict__ rpb, h16* __restrict__ aoh) {
  constexpr int XS = 200, QS = 40, VS = 72, PS = 72;
  __shared__ h16  xw_s[NPAD * XS];        // 25.0 KB, reused by probs
  __shared__ h16  qs_s[NPAD * QS];        // 5.0 KB
  __shared__ h16  ks_s[NPAD * QS];        // 5.0 KB
  __shared__ h16  vt_s[HD * VS];          // 4.5 KB (v^T: [d][m])
  __shared__ float sc_s[NPAD * NPAD];     // 16 KB
  h16* probs = xw_s;

  int win = blockIdx.x / HEADS;
  int h   = blockIdx.x % HEADS;
  int widx = win & 63;
  int wh = widx >> 3, ww = widx & 7;
  int tid = threadIdx.x;
  int wave = __builtin_amdgcn_readfirstlane(tid >> 5);   // wave-uniform -> SGPR
  int lane = tid & 31;

  const h16* src = xwh + win * (NPAD * DIM);
  __builtin_prefetch(wq + (h * 2) * 6 * 512, 0, 0);      // warm this head's weight slice
  for (int i = tid; i < (NPAD * DIM) / 8; i += 256) {
    int f = i * 8, n = f / DIM, c = f % DIM;
    async_copy16(src + f, xw_s + n * XS + c);
  }
  async_wait();
  __syncthreads();

  // QKV GEMM for this head: 64x96 output, K=192 (24 tiles / 8 waves)
  for (int i = 0; i < 3; ++i) {
    int tile = wave * 3 + i;
    int mt = tile & 3, oi = tile >> 2;
    int which = oi >> 1, db16 = oi & 1;            // 0=q 1=k 2=v
    int ot = which * 12 + h * 2 + db16;            // tile index into swizzled qkv_w
    v8f acc = {};
    for (int kc = 0; kc < 6; ++kc) {
      v16h a = lds_load_a(xw_s, XS, mt * 16, kc * 32, lane);
      v16h b = gbl_load_w(wq, 6, ot, kc, lane);
      acc = wmma16(a, b, acc);
    }
    int col = lane & 15, hi = (lane >> 4) & 1;
    int d = db16 * 16 + col;
    float bias = qkv_b[which * DIM + h * HD + d];
    for (int v = 0; v < 8; ++v) {
      int m = mt * 16 + v + 8 * hi;
      h16 hv = (h16)(acc[v] + bias);
      if (which == 0)      qs_s[m * QS + d] = hv;  // scalar branch: which is SGPR
      else if (which == 1) ks_s[m * QS + d] = hv;
      else                 vt_s[d * VS + m] = hv;
    }
  }
  __syncthreads();

  // cosine normalize rows; fold logit scale into q
  float sc_h = expf(fminf(logit_scale[h], 4.6051702f));
  for (int t = tid; t < 2 * NTOK; t += 256) {
    bool isq = t < NTOK;
    int m = isq ? t : t - NTOK;
    h16* row = (isq ? qs_s : ks_s) + m * QS;
    float ss = 0.f;
    for (int d = 0; d < HD; ++d) { float f = (float)row[d]; ss += f * f; }
    float inv = 1.0f / fmaxf(sqrtf(ss), 1e-12f);
    if (isq) inv *= sc_h;
    for (int d = 0; d < HD; ++d) row[d] = (h16)((float)row[d] * inv);
  }
  __syncthreads();

  // scores = qn @ kn^T  (64x64, K=32): 16 tiles / 8 waves
  for (int i = 0; i < 2; ++i) {
    int tile = wave * 2 + i;
    int mt = tile >> 2, nt = tile & 3;
    v16h a = lds_load_a(qs_s, QS, mt * 16, 0, lane);
    v16h b = lds_load_bt(ks_s, QS, nt * 16, 0, lane);
    v8f acc = {};
    acc = wmma16(a, b, acc);
    int col = nt * 16 + (lane & 15), hi = (lane >> 4) & 1;
    for (int v = 0; v < 8; ++v)
      sc_s[(mt * 16 + v + 8 * hi) * NPAD + col] = acc[v];
  }
  __syncthreads();

  // bias + shifted-window mask + softmax -> probs f16 (aliases xw_s)
  for (int r = tid; r < NPAD; r += 256) {
    if (r < NTOK) {
      int rh = wh * WS_ + r / WS_, rw = ww * WS_ + r % WS_;
      int rgr = (rh < 49 ? 0 : (rh < 53 ? 1 : 2)) * 3 + (rw < 49 ? 0 : (rw < 53 ? 1 : 2));
      float mx = -1e30f;
      for (int j = 0; j < NTOK; ++j) {
        int gh = wh * WS_ + j / WS_, gw = ww * WS_ + j % WS_;
        int rgj = (gh < 49 ? 0 : (gh < 53 ? 1 : 2)) * 3 + (gw < 49 ? 0 : (gw < 53 ? 1 : 2));
        float l = sc_s[r * NPAD + j] + rpb[(h * NTOK + r) * NTOK + j] + (rgj == rgr ? 0.f : -100.f);
        sc_s[r * NPAD + j] = l;
        mx = fmaxf(mx, l);
      }
      float sum = 0.f;
      for (int j = 0; j < NTOK; ++j) {
        float e = expf(sc_s[r * NPAD + j] - mx);
        sc_s[r * NPAD + j] = e; sum += e;
      }
      float inv = 1.f / sum;
      for (int j = 0; j < NPAD; ++j)
        probs[r * PS + j] = (j < NTOK) ? (h16)(sc_s[r * NPAD + j] * inv) : (h16)0.f;
    } else {
      for (int j = 0; j < NPAD; ++j) probs[r * PS + j] = (h16)0.f;
    }
  }
  __syncthreads();

  // out = probs @ v  (64x32, K=64): 8 tiles -> one per wave
  {
    int mt = wave >> 1, dt = wave & 1;
    v8f acc = {};
    for (int kc = 0; kc < 2; ++kc) {
      v16h a = lds_load_a(probs, PS, mt * 16, kc * 32, lane);
      v16h b = lds_load_bt(vt_s, VS, dt * 16, kc * 32, lane);
      acc = wmma16(a, b, acc);
    }
    int col = lane & 15, hi = (lane >> 4) & 1;
    int d = dt * 16 + col;
    h16* dst = aoh + win * (NPAD * DIM);
    for (int v = 0; v < 8; ++v) {
      int m = mt * 16 + v + 8 * hi;
      dst[m * DIM + h * HD + d] = (h16)acc[v];
    }
  }
}

// ---- proj GEMM + LayerNorm1 + shortcut residual (window reverse + unshift) ----
__global__ __launch_bounds__(256) void k_proj(
    const h16* __restrict__ aoh, const h16* __restrict__ wp,
    const float* __restrict__ proj_b, const float* __restrict__ x,
    const float* __restrict__ n1w, const float* __restrict__ n1b,
    float* __restrict__ x1, h16* __restrict__ x1h) {
  constexpr int AS = 200;
  __shared__ h16 ao_s[NPAD * AS];
  __shared__ h16 out_s[NPAD * AS];
  int win = blockIdx.x;
  int tid = threadIdx.x;
  int wave = __builtin_amdgcn_readfirstlane(tid >> 5);
  int lane = tid & 31;

  const h16* src = aoh + win * (NPAD * DIM);
  __builtin_prefetch(wp, 0, 0);
  for (int i = tid; i < (NPAD * DIM) / 8; i += 256) {
    int f = i * 8, n = f / DIM, c = f % DIM;
    async_copy16(src + f, ao_s + n * AS + c);
  }
  async_wait();
  __syncthreads();

  for (int i = 0; i < 6; ++i) {                  // 48 tiles / 8 waves
    int tile = wave * 6 + i;
    int mt = tile & 3, ot = tile >> 2;
    v8f acc = {};
    for (int kc = 0; kc < 6; ++kc) {
      v16h a = lds_load_a(ao_s, AS, mt * 16, kc * 32, lane);
      v16h b = gbl_load_w(wp, 6, ot, kc, lane);
      acc = wmma16(a, b, acc);
    }
    int col = ot * 16 + (lane & 15), hi = (lane >> 4) & 1;
    float bias = proj_b[col];
    for (int v = 0; v < 8; ++v)
      out_s[(mt * 16 + v + 8 * hi) * AS + col] = (h16)(acc[v] + bias);
  }
  __syncthreads();

  int b = win >> 6, widx = win & 63, wh = widx >> 3, ww = widx & 7;
  for (int r = tid; r < NTOK; r += 256) {
    int oh = (wh * WS_ + r / WS_ + SS_) % IMG;
    int ow = (ww * WS_ + r % WS_ + SS_) % IMG;
    long tok = (long)(b * IMG + oh) * IMG + ow;
    float mean = 0.f;
    for (int c = 0; c < DIM; ++c) mean += (float)out_s[r * AS + c];
    mean *= (1.0f / DIM);
    float var = 0.f;
    for (int c = 0; c < DIM; ++c) { float d = (float)out_s[r * AS + c] - mean; var += d * d; }
    var *= (1.0f / DIM);
    float inv = rsqrtf(var + 1e-5f);
    for (int c = 0; c < DIM; ++c) {
      float v = ((float)out_s[r * AS + c] - mean) * inv * n1w[c] + n1b[c] + x[tok * DIM + c];
      x1[tok * DIM + c]  = v;
      x1h[tok * DIM + c] = (h16)v;
    }
  }
}

// ---- fused MLP: gelu(x1@fc1^T+b1)@fc2^T+b2, LayerNorm2, residual ----
__global__ __launch_bounds__(256) void k_mlp(
    const h16* __restrict__ x1h, const h16* __restrict__ w1,
    const float* __restrict__ fc1_b, const h16* __restrict__ w2,
    const float* __restrict__ fc2_b, const float* __restrict__ x1,
    const float* __restrict__ n2w, const float* __restrict__ n2b,
    float* __restrict__ out) {
  constexpr int XS = 200, HS = 264;
  __shared__ h16 x1_s[NPAD * XS];   // 25.0 KB, reused for y
  __shared__ h16 h_s[NPAD * HS];    // 33.0 KB (256-wide hidden chunk)
  h16* y_s = x1_s;
  int tb = blockIdx.x;
  int tid = threadIdx.x;
  int wave = __builtin_amdgcn_readfirstlane(tid >> 5);
  int lane = tid & 31;

  const h16* src = x1h + (long)tb * NPAD * DIM;
  __builtin_prefetch(w1, 0, 0);
  __builtin_prefetch(w2, 0, 0);
  for (int i = tid; i < (NPAD * DIM) / 8; i += 256) {
    int f = i * 8, n = f / DIM, c = f % DIM;
    async_copy16(src + f, x1_s + n * XS + c);
  }
  async_wait();
  __syncthreads();

  v8f zero = {};
  v8f yacc[6];
  for (int i = 0; i < 6; ++i) yacc[i] = zero;

  for (int p = 0; p < 3; ++p) {
    // GEMM1 chunk: hidden cols [p*256, p*256+256); 64 tiles / 8 waves
    for (int i = 0; i < 8; ++i) {
      int tile = wave + 8 * i;
      int mt = tile & 3, hot_l = tile >> 2;
      int hot = p * 16 + hot_l;
      v8f acc = {};
      for (int kc = 0; kc < 6; ++kc) {
        v16h a = lds_load_a(x1_s, XS, mt * 16, kc * 32, lane);
        v16h b = gbl_load_w(w1, 6, hot, kc, lane);
        acc = wmma16(a, b, acc);
      }
      int col = lane & 15, hi = (lane >> 4) & 1;
      float bias = fc1_b[hot * 16 + col];
      for (int v = 0; v < 8; ++v) {
        float g = acc[v] + bias;
        g = 0.5f * g * (1.0f + erff(g * 0.70710678f));     // exact GELU
        h_s[(mt * 16 + v + 8 * hi) * HS + hot_l * 16 + col] = (h16)g;
      }
    }
    __syncthreads();
    // GEMM2 partial accumulate: y += h_chunk @ fc2_chunk^T (48 tiles / 8 waves)
    for (int i = 0; i < 6; ++i) {
      int tile = wave * 6 + i;
      int mt = tile & 3, ot = tile >> 2;
      for (int kc = 0; kc < 8; ++kc) {
        v16h a = lds_load_a(h_s, HS, mt * 16, kc * 32, lane);
        v16h b = gbl_load_w(w2, 24, ot, p * 8 + kc, lane);
        yacc[i] = wmma16(a, b, yacc[i]);
      }
    }
    __syncthreads();
  }

  // spill y (+fc2_b) to LDS (aliases x1_s; all x1 reads completed)
  for (int i = 0; i < 6; ++i) {
    int tile = wave * 6 + i;
    int mt = tile & 3, ot = tile >> 2;
    int col = ot * 16 + (lane & 15), hi = (lane >> 4) & 1;
    float bias = fc2_b[col];
    for (int v = 0; v < 8; ++v)
      y_s[(mt * 16 + v + 8 * hi) * XS + col] = (h16)(yacc[i][v] + bias);
  }
  __syncthreads();

  for (int r = tid; r < NPAD; r += 256) {
    long tok = (long)tb * NPAD + r;
    float mean = 0.f;
    for (int c = 0; c < DIM; ++c) mean += (float)y_s[r * XS + c];
    mean *= (1.0f / DIM);
    float var = 0.f;
    for (int c = 0; c < DIM; ++c) { float d = (float)y_s[r * XS + c] - mean; var += d * d; }
    var *= (1.0f / DIM);
    float inv = rsqrtf(var + 1e-5f);
    for (int c = 0; c < DIM; ++c)
      out[tok * DIM + c] = ((float)y_s[r * XS + c] - mean) * inv * n2w[c] + n2b[c] + x1[tok * DIM + c];
  }
}

extern "C" void kernel_launch(void* const* d_in, const int* in_sizes, int n_in,
                              void* d_out, int out_size, void* d_ws, size_t ws_size,
                              hipStream_t stream) {
  (void)in_sizes; (void)n_in; (void)out_size; (void)ws_size;
  const float* x       = (const float*)d_in[0];
  const float* qkv_w   = (const float*)d_in[1];
  const float* qkv_b   = (const float*)d_in[2];
  const float* proj_w  = (const float*)d_in[3];
  const float* proj_b  = (const float*)d_in[4];
  const float* logit_s = (const float*)d_in[5];
  const float* cpb_w1  = (const float*)d_in[6];
  const float* cpb_b1  = (const float*)d_in[7];
  const float* cpb_w2  = (const float*)d_in[8];
  const float* n1w     = (const float*)d_in[9];
  const float* n1b     = (const float*)d_in[10];
  const float* n2w     = (const float*)d_in[11];
  const float* n2b     = (const float*)d_in[12];
  const float* fc1_w   = (const float*)d_in[13];
  const float* fc1_b   = (const float*)d_in[14];
  const float* fc2_w   = (const float*)d_in[15];
  const float* fc2_b   = (const float*)d_in[16];
  float* out = (float*)d_out;

  char* ws = (char*)d_ws;
  size_t off = 0;
  auto alloc = [&](size_t bytes) {
    void* p = ws + off;
    off = (off + bytes + 255) & ~(size_t)255;
    return p;
  };
  float* rpb = (float*)alloc((size_t)HEADS * NTOK * NTOK * sizeof(float));
  h16* xwh = (h16*)alloc((size_t)NWINDOWS * NPAD * DIM * sizeof(h16));
  h16* aoh = (h16*)alloc((size_t)NWINDOWS * NPAD * DIM * sizeof(h16));
  float* x1 = (float*)alloc((size_t)TOKENS * DIM * sizeof(float));
  h16* x1h = (h16*)alloc((size_t)TOKENS * DIM * sizeof(h16));
  h16* wq = (h16*)alloc((size_t)576 * 192 * sizeof(h16));
  h16* wp = (h16*)alloc((size_t)192 * 192 * sizeof(h16));
  h16* w1 = (h16*)alloc((size_t)768 * 192 * sizeof(h16));
  h16* w2 = (h16*)alloc((size_t)192 * 768 * sizeof(h16));

  k_wconv<<<(36 * 6 * 512 + 255) / 256, 256, 0, stream>>>(qkv_w, wq, 36, 6, 192);
  k_wconv<<<(12 * 6 * 512 + 255) / 256, 256, 0, stream>>>(proj_w, wp, 12, 6, 192);
  k_wconv<<<(48 * 6 * 512 + 255) / 256, 256, 0, stream>>>(fc1_w, w1, 48, 6, 192);
  k_wconv<<<(12 * 24 * 512 + 255) / 256, 256, 0, stream>>>(fc2_w, w2, 12, 24, 768);
  k_bias<<<1, 256, 0, stream>>>(cpb_w1, cpb_b1, cpb_w2, rpb);
  k_pack<<<NWINDOWS, 256, 0, stream>>>(x, xwh);
  k_attn<<<NWINDOWS * HEADS, 256, 0, stream>>>(xwh, wq, qkv_b, logit_s, rpb, aoh);
  k_proj<<<NWINDOWS, 256, 0, stream>>>(aoh, wp, proj_b, x, n1w, n1b, x1, x1h);
  k_mlp<<<TOKENS / NPAD, 256, 0, stream>>>(x1h, w1, fc1_b, w2, fc2_b, x1, n2w, n2b, out);
}
